// GraphAttentionHead_9964324127022
// MI455X (gfx1250) — compile-verified, matrix operand order checked by reading, not run on hardware
//
#include <hip/hip_runtime.h>

// ---------------------------------------------------------------------------
// Types (pure ext_vector types: trivially constructible, union-safe)
// ---------------------------------------------------------------------------
typedef __attribute__((ext_vector_type(4)))  unsigned int u32x4;
typedef __attribute__((ext_vector_type(2)))  unsigned int u32x2;
typedef __attribute__((ext_vector_type(4)))  float        f32x4;
typedef __attribute__((ext_vector_type(16))) _Float16     v16h;
typedef __attribute__((ext_vector_type(8)))  float        v8f;

union Frag { u32x4 u[2]; v16h v; };
static_assert(sizeof(Frag) == 32, "frag is 8 VGPRs");

__device__ __forceinline__ v8f wmma_f16(const Frag& a, const Frag& b, v8f c) {
    // D = A(16x32 f16) * B(32x16 f16) + C(16x16 f32)
    return __builtin_amdgcn_wmma_f32_16x16x32_f16(
        /*neg_a=*/false, a.v, /*neg_b=*/false, b.v,
        /*c_mod=*/(short)0, c, /*reuse_a=*/false, /*reuse_b=*/false);
}

// CDNA5 async global->LDS copy (ASYNCcnt-tracked, bypasses VGPRs).
// lds_off = low 32 bits of the flat LDS address (LDS aperture: ADDR[31:0]).
__device__ __forceinline__ void async_ldg_b128(unsigned int lds_off, const void* gptr) {
    asm volatile("global_load_async_to_lds_b128 %0, %1, off"
                 :: "v"(lds_off), "v"(gptr) : "memory");
}
__device__ __forceinline__ void wait_async0() {
    asm volatile("s_wait_asynccnt 0x0" ::: "memory");
}

#define N_TOK 8192
#define DDIM  1024
#define NEMB  102   // MAX_DIST + 2

// ---------------------------------------------------------------------------
// f32 -> f16 elementwise convert (4 elems / thread, b128 in, b64 out)
// ---------------------------------------------------------------------------
__global__ void k_cvt_f16(const float* __restrict__ in, _Float16* __restrict__ out) {
    size_t i4 = (size_t)blockIdx.x * blockDim.x + threadIdx.x;
    f32x4 f = ((const f32x4*)in)[i4];
    union { _Float16 h[4]; u32x2 u; } pk;
    pk.h[0] = (_Float16)f[0]; pk.h[1] = (_Float16)f[1];
    pk.h[2] = (_Float16)f[2]; pk.h[3] = (_Float16)f[3];
    ((u32x2*)out)[i4] = pk.u;
}

// ---------------------------------------------------------------------------
// Tiled transpose f32[rows][cols] -> f16[cols][rows], with scale
// ---------------------------------------------------------------------------
__global__ void k_transpose_f32_f16(const float* __restrict__ in, _Float16* __restrict__ out,
                                    int rows, int cols, float scale) {
    __shared__ float t[32][33];
    const int bx = blockIdx.x * 32, by = blockIdx.y * 32;
    const int tx = threadIdx.x, ty = threadIdx.y;
#pragma unroll
    for (int j = 0; j < 32; j += 8)
        t[ty + j][tx] = in[(size_t)(by + ty + j) * cols + (bx + tx)];
    __syncthreads();
#pragma unroll
    for (int j = 0; j < 32; j += 8)
        out[(size_t)(bx + ty + j) * rows + (by + tx)] = (_Float16)(t[tx][ty + j] * scale);
}

// Tiled transpose f16[rows][cols] -> f16[cols][rows]
__global__ void k_transpose_f16(const _Float16* __restrict__ in, _Float16* __restrict__ out,
                                int rows, int cols) {
    __shared__ _Float16 t[32][33];
    const int bx = blockIdx.x * 32, by = blockIdx.y * 32;
    const int tx = threadIdx.x, ty = threadIdx.y;
#pragma unroll
    for (int j = 0; j < 32; j += 8)
        t[ty + j][tx] = in[(size_t)(by + ty + j) * cols + (bx + tx)];
    __syncthreads();
#pragma unroll
    for (int j = 0; j < 32; j += 8)
        out[(size_t)(bx + ty + j) * rows + (by + tx)] = t[tx][ty + j];
}

// ---------------------------------------------------------------------------
// WMMA GEMM:  C[M][Nn] = A[M][K] * Bt[Nn][K]^T   (both operands f16 row-major,
// Bt pre-transposed so A and B fragments use the identical load pattern).
// Block = 256 thr (8 waves), tile 64x64, k-step 32.
// LDS staging via CDNA5 GLOBAL_LOAD_ASYNC_TO_LDS_B128, double-buffered:
// tile k+1 streams into the other buffer while WMMAs consume tile k; one
// s_wait_asynccnt + one barrier per k-step.
// ---------------------------------------------------------------------------
template <bool OUT_F32>
__global__ __launch_bounds__(256, 2)
void k_gemm(const _Float16* __restrict__ A, const _Float16* __restrict__ Bt,
            void* __restrict__ Cout, int M, int Nn, int K) {
    __shared__ alignas(16) _Float16 As[2][64 * 32];
    __shared__ alignas(16) _Float16 Bs[2][64 * 32];

    const int tid  = threadIdx.x;
    const int m0   = blockIdx.y * 64, n0 = blockIdx.x * 64;
    const int frow = tid >> 2, fcol = (tid & 3) << 3;   // staging: 64 rows x 32 cols
    const int wid  = tid >> 5, lane = tid & 31;
    const int half = lane >> 4, r16 = lane & 15;
    const int wm   = wid >> 1, wn = wid & 1;            // 4x2 wave grid -> 16x32 per wave

    v8f c0 = {0.f,0.f,0.f,0.f,0.f,0.f,0.f,0.f};
    v8f c1 = c0;

    const _Float16* aG = A  + (size_t)(m0 + frow) * K + fcol;
    const _Float16* bG = Bt + (size_t)(n0 + frow) * K + fcol;
    const unsigned int aL[2] = { (unsigned int)(uintptr_t)&As[0][frow * 32 + fcol],
                                 (unsigned int)(uintptr_t)&As[1][frow * 32 + fcol] };
    const unsigned int bL[2] = { (unsigned int)(uintptr_t)&Bs[0][frow * 32 + fcol],
                                 (unsigned int)(uintptr_t)&Bs[1][frow * 32 + fcol] };

    // prologue: stage tile 0 into buffer 0
    async_ldg_b128(aL[0], aG);
    async_ldg_b128(bL[0], bG);

    int buf = 0;
    for (int k0 = 0; k0 < K; k0 += 32) {
        wait_async0();          // this wave's async LDS writes done
        __syncthreads();        // => everyone's writes done + prev reads retired

        if (k0 + 32 < K) {      // stream next tile into the other buffer
            async_ldg_b128(aL[buf ^ 1], aG + (k0 + 32));
            async_ldg_b128(bL[buf ^ 1], bG + (k0 + 32));
        }

        Frag a, b0, b1;
        const char* ap  = (const char*)&As[buf][(wm * 16 + r16) * 32] + half * 16;
        a.u[0]  = *(const u32x4*)ap;        a.u[1]  = *(const u32x4*)(ap + 32);
        const char* bp0 = (const char*)&Bs[buf][(wn * 32 + r16) * 32] + half * 16;
        b0.u[0] = *(const u32x4*)bp0;       b0.u[1] = *(const u32x4*)(bp0 + 32);
        const char* bp1 = bp0 + 16 * 32 * 2;    // +16 B-rows
        b1.u[0] = *(const u32x4*)bp1;       b1.u[1] = *(const u32x4*)(bp1 + 32);

        c0 = wmma_f16(a, b0, c0);
        c1 = wmma_f16(a, b1, c1);
        buf ^= 1;
    }

    // C layout: VGPR r -> row (r + 8*half), col = lane%16
    const int mB = m0 + wm * 16 + half * 8;
    const int nC = n0 + wn * 32 + r16;
#pragma unroll
    for (int r = 0; r < 8; ++r) {
        const size_t o0 = (size_t)(mB + r) * Nn + nC;
        if constexpr (OUT_F32) {
            ((float*)Cout)[o0]      = c0[r];
            ((float*)Cout)[o0 + 16] = c1[r];
        } else {
            ((_Float16*)Cout)[o0]      = (_Float16)c0[r];
            ((_Float16*)Cout)[o0 + 16] = (_Float16)c1[r];
        }
    }
}

// ---------------------------------------------------------------------------
// Scores pass:  S[i][j] = (q_i . k_j) (scale pre-folded into Q) + emb[dist[i][j]]
// Writes raw S (f32) once, and per-row online-softmax stats (m, l).
// Wave owns 16 query rows; Q fragments register-resident (256 VGPRs);
// K streamed from global (L2-resident, 16 MB).
// ---------------------------------------------------------------------------
__global__ __launch_bounds__(256, 1)
void k_scores(const _Float16* __restrict__ Q, const _Float16* __restrict__ Kmat,
              const int* __restrict__ dist, const float* __restrict__ embg,
              float* __restrict__ S, float* __restrict__ mOut, float* __restrict__ lOut) {
    __shared__ float emb[NEMB];
    const int tid = threadIdx.x;
    if (tid < NEMB) emb[tid] = embg[tid];
    __syncthreads();

    const int wid  = tid >> 5, lane = tid & 31;
    const int half = lane >> 4, r16 = lane & 15;
    const int qbase = blockIdx.x * 128 + wid * 16;

    // Preload all 32 Q fragments for this wave's 16 rows (A-matrix layout).
    Frag qf[32];
    const char* qrow = (const char*)(Q + (size_t)(qbase + r16) * DDIM) + half * 16;
#pragma unroll
    for (int c = 0; c < 32; ++c) {
        qf[c].u[0] = *(const u32x4*)(qrow + c * 64);
        qf[c].u[1] = *(const u32x4*)(qrow + c * 64 + 32);
    }

    float mrun[8], lrun[8];
#pragma unroll
    for (int r = 0; r < 8; ++r) { mrun[r] = -3.0e38f; lrun[r] = 0.f; }

    for (int j0 = 0; j0 < N_TOK; j0 += 32) {
        v8f c0 = {0.f,0.f,0.f,0.f,0.f,0.f,0.f,0.f};
        v8f c1 = c0;
        const char* k0p = (const char*)(Kmat + (size_t)(j0 + r16) * DDIM) + half * 16;
        const char* k1p = (const char*)(Kmat + (size_t)(j0 + 16 + r16) * DDIM) + half * 16;
        // prefetch this lane's K rows of the next key tile (32 rows ahead)
        __builtin_prefetch(k0p + (size_t)32 * DDIM * 2, 0, 3);
        __builtin_prefetch(k1p + (size_t)32 * DDIM * 2, 0, 3);
#pragma unroll
        for (int c = 0; c < 32; ++c) {
            Frag b0, b1;
            b0.u[0] = *(const u32x4*)(k0p + c * 64);
            b0.u[1] = *(const u32x4*)(k0p + c * 64 + 32);
            b1.u[0] = *(const u32x4*)(k1p + c * 64);
            b1.u[1] = *(const u32x4*)(k1p + c * 64 + 32);
            c0 = wmma_f16(qf[c], b0, c0);
            c1 = wmma_f16(qf[c], b1, c1);
        }

        const int col0 = j0 + r16, col1 = col0 + 16;
#pragma unroll
        for (int r = 0; r < 8; ++r) {
            const size_t roff = (size_t)(qbase + half * 8 + r) * (size_t)N_TOK;
            float s0 = c0[r] + emb[dist[roff + col0]];
            float s1 = c1[r] + emb[dist[roff + col1]];
            S[roff + col0] = s0;
            S[roff + col1] = s1;
            // row-wise tile max across the 16 lanes holding this row
            float t = fmaxf(s0, s1);
#pragma unroll
            for (int o = 1; o < 16; o <<= 1) t = fmaxf(t, __shfl_xor(t, o, 32));
            const float mo = mrun[r];
            const float mn = fmaxf(mo, t);
            float ps = __expf(s0 - mn) + __expf(s1 - mn);
#pragma unroll
            for (int o = 1; o < 16; o <<= 1) ps += __shfl_xor(ps, o, 32);
            lrun[r] = lrun[r] * __expf(mo - mn) + ps;   // online rescale of running sum
            mrun[r] = mn;
        }
    }

    if (r16 == 0) {
#pragma unroll
        for (int r = 0; r < 8; ++r) {
            const int row = qbase + half * 8 + r;
            mOut[row] = mrun[r];
            lOut[row] = lrun[r];
        }
    }
}

// ---------------------------------------------------------------------------
// Normalize: P = f16( exp(S - m_row) / l_row ), 4 elems / thread
// ---------------------------------------------------------------------------
__global__ void k_softmax_norm(const float* __restrict__ S, const float* __restrict__ mIn,
                               const float* __restrict__ lIn, _Float16* __restrict__ P) {
    const size_t i4  = (size_t)blockIdx.x * blockDim.x + threadIdx.x;
    const size_t base = i4 * 4;
    const int row = (int)(base >> 13);                  // /8192, groups never cross rows
    const f32x4 s = ((const f32x4*)S)[i4];
    const float mm = mIn[row];
    const float rl = 1.0f / lIn[row];
    union { _Float16 h[4]; u32x2 u; } pk;
    pk.h[0] = (_Float16)(__expf(s[0] - mm) * rl);
    pk.h[1] = (_Float16)(__expf(s[1] - mm) * rl);
    pk.h[2] = (_Float16)(__expf(s[2] - mm) * rl);
    pk.h[3] = (_Float16)(__expf(s[3] - mm) * rl);
    ((u32x2*)P)[i4] = pk.u;
}

// ---------------------------------------------------------------------------
// Host-side orchestration
// ---------------------------------------------------------------------------
extern "C" void kernel_launch(void* const* d_in, const int* in_sizes, int n_in,
                              void* d_out, int out_size, void* d_ws, size_t ws_size,
                              hipStream_t stream) {
    (void)in_sizes; (void)n_in; (void)out_size; (void)ws_size;

    const float* X    = (const float*)d_in[0];
    const int*   dist = (const int*)d_in[1];
    const float* Wq   = (const float*)d_in[2];
    const float* Wk   = (const float*)d_in[3];
    const float* Wv   = (const float*)d_in[4];
    const float* emb  = (const float*)d_in[5];
    float*       out  = (float*)d_out;

    const size_t NX = (size_t)N_TOK * DDIM;     // 8.39M
    const size_t NW = (size_t)DDIM * DDIM;      // 1.05M
    const size_t NS = (size_t)N_TOK * N_TOK;    // 67.1M

    char* ws = (char*)d_ws;
    size_t off = 0;
    auto carve = [&](size_t bytes) -> void* {
        void* p = ws + off;
        off = (off + bytes + 255) & ~(size_t)255;
        return p;
    };
    _Float16* Xh   = (_Float16*)carve(NX * 2);
    _Float16* Wqt  = (_Float16*)carve(NW * 2);
    _Float16* Wkt  = (_Float16*)carve(NW * 2);
    _Float16* Wvt  = (_Float16*)carve(NW * 2);
    _Float16* Qh   = (_Float16*)carve(NX * 2);
    _Float16* Kh   = (_Float16*)carve(NX * 2);
    _Float16* Vh   = (_Float16*)carve(NX * 2);
    _Float16* Vt   = (_Float16*)carve(NX * 2);
    float*    Sbuf = (float*)carve(NS * 4);
    _Float16* Pbuf = (_Float16*)carve(NS * 2);
    float*    mb   = (float*)carve((size_t)N_TOK * 4);
    float*    lb   = (float*)carve((size_t)N_TOK * 4);

    // 1) X -> f16
    k_cvt_f16<<<dim3((unsigned)(NX / 1024)), dim3(256), 0, stream>>>(X, Xh);

    // 2) W^T -> f16 (fold 1/sqrt(D)=1/32 score scale into W_q)
    dim3 tb(32, 8);
    k_transpose_f32_f16<<<dim3(32, 32), tb, 0, stream>>>(Wq, Wqt, DDIM, DDIM, 0.03125f);
    k_transpose_f32_f16<<<dim3(32, 32), tb, 0, stream>>>(Wk, Wkt, DDIM, DDIM, 1.0f);
    k_transpose_f32_f16<<<dim3(32, 32), tb, 0, stream>>>(Wv, Wvt, DDIM, DDIM, 1.0f);

    // 3) Q/K/V projections (WMMA GEMM, f16 out)
    dim3 gqkv(DDIM / 64, N_TOK / 64);   // (16, 128)
    k_gemm<false><<<gqkv, dim3(256), 0, stream>>>(Xh, Wqt, Qh, N_TOK, DDIM, DDIM);
    k_gemm<false><<<gqkv, dim3(256), 0, stream>>>(Xh, Wkt, Kh, N_TOK, DDIM, DDIM);
    k_gemm<false><<<gqkv, dim3(256), 0, stream>>>(Xh, Wvt, Vh, N_TOK, DDIM, DDIM);

    // 4) V^T for the P@V GEMM B-operand
    k_transpose_f16<<<dim3(DDIM / 32, N_TOK / 32), tb, 0, stream>>>(Vh, Vt, N_TOK, DDIM);

    // 5) scores + online softmax stats (dist read exactly once)
    k_scores<<<dim3(N_TOK / 128), dim3(256), 0, stream>>>(Qh, Kh, dist, emb, Sbuf, mb, lb);

    // 6) P = softmax(S) in f16
    k_softmax_norm<<<dim3((unsigned)(NS / 1024)), dim3(256), 0, stream>>>(Sbuf, mb, lb, Pbuf);

    // 7) out = P @ V  (WMMA GEMM, f32 out)
    k_gemm<true><<<gqkv, dim3(256), 0, stream>>>(Pbuf, Vt, out, N_TOK, DDIM, N_TOK);
}
